// CrossViewAttention_48790828482898
// MI455X (gfx1250) — matrix operand order repaired and checked by reading, not compiled, not run on hardware
//
#include <hip/hip_runtime.h>
#include <hip/hip_bf16.h>

typedef __attribute__((ext_vector_type(16))) __bf16 v16bf;
typedef __attribute__((ext_vector_type(8)))  __bf16 v8bf;
typedef __attribute__((ext_vector_type(8)))  float  v8f;
typedef __attribute__((ext_vector_type(4)))  unsigned int u32x4;
typedef __attribute__((ext_vector_type(8)))  int i32x8;
typedef __attribute__((ext_vector_type(4)))  int i32x4;

constexpr int NVIEW = 6;
constexpr int BV    = 2;
constexpr int NB    = NVIEW * BV;   // 12 flattened batches
constexpr int CH    = 512;          // channels
constexpr int TT    = 1024;         // H*W tokens
constexpr int NHEAD = 8;
constexpr int HD    = CH / NHEAD;   // 64

static __device__ __forceinline__ v8f wmma_bf16(v16bf a, v16bf b, v8f c) {
  return __builtin_amdgcn_wmma_f32_16x16x32_bf16(false, a, false, b, (short)0, c,
                                                 false, false);
}

// A-fragment (16x32 bf16): lane L holds row M=L%16; half h=L/16 holds
// K in [8h,8h+8) and [16+8h,16+8h+8)  (ISA 7.12.2).
static __device__ __forceinline__ v16bf load_a_frag(const __bf16* base, int pitch,
                                                    int row, int k0, int lane) {
  const int m = lane & 15, h = lane >> 4;
  const __bf16* p = base + (size_t)(row + m) * pitch + k0;
  v16bf r;
#pragma unroll
  for (int i = 0; i < 8; ++i) r[i] = p[h * 8 + i];
#pragma unroll
  for (int i = 0; i < 8; ++i) r[8 + i] = p[16 + h * 8 + i];
  return r;
}

// B-fragment (32x16 bf16) from LDS stored transposed [n][k]:
// lane L holds col n=L%16; half h holds K in [8h,8h+8) and [16+8h,16+8h+8).
static __device__ __forceinline__ v16bf load_b_frag(const __bf16* baseT, int pitch,
                                                    int col, int k0, int lane) {
  const int n = lane & 15, h = lane >> 4;
  const __bf16* p = baseT + (size_t)(col + n) * pitch + k0 + h * 8;
  v16bf r;
#pragma unroll
  for (int i = 0; i < 8; ++i) r[i] = p[i];
#pragma unroll
  for (int i = 0; i < 8; ++i) r[8 + i] = p[16 + i];
  return r;
}

// CDNA5 async direct-to-LDS copy (ASYNCcnt); vdst = wave-relative LDS byte
// address, vaddr = 64-bit global address.
static __device__ __forceinline__ void async_load_b128(unsigned lds_addr,
                                                       const void* gaddr) {
  unsigned long long ga = (unsigned long long)gaddr;
  asm volatile("global_load_async_to_lds_b128 %0, %1, off"
               :: "v"(lds_addr), "v"(ga) : "memory");
}
static __device__ __forceinline__ void wait_asynccnt0() {
  asm volatile("s_wait_asynccnt 0" ::: "memory");
}
static __device__ __forceinline__ unsigned lds_off(const void* p) {
  return (unsigned)(unsigned long long)p;  // low 32 bits of generic LDS ptr
}

// ---------------------------------------------------------------------------
// Kernel 1: KV projection (needs f32->bf16 conversion on both operands, so
// staging stays on the VALU path).  Block tile 128x64, 8 waves x 32x32.
// ---------------------------------------------------------------------------
constexpr int G_TM = 128, G_TN = 64, G_TK = 32;
constexpr int G_AP = 40, G_BP = 40;

__global__ __launch_bounds__(256) void kv_proj_kernel(
    const float* __restrict__ kv, const float* __restrict__ Wkv,
    const float* __restrict__ bkv, __bf16* __restrict__ Kbuf,
    __bf16* __restrict__ Vbuf) {
  __shared__ __align__(16) __bf16 As[G_TM * G_AP];
  __shared__ __align__(16) __bf16 Bt[G_TN * G_BP];

  const int tid = threadIdx.x, lane = tid & 31, wave = tid >> 5;
  const int wm = wave & 3, wn = wave >> 2;
  const int m0 = blockIdx.x * G_TM;
  const int n0 = blockIdx.y * G_TN;
  const int bb = m0 / TT, t0 = m0 % TT;

  v8f acc[2][2];
  v8f zero = {};
#pragma unroll
  for (int i = 0; i < 2; ++i)
#pragma unroll
    for (int j = 0; j < 2; ++j) acc[i][j] = zero;

  for (int kt = 0; kt < CH / G_TK; ++kt) {
#pragma unroll
    for (int it = 0; it < (G_TM * G_TK) / 256; ++it) {
      int idx = tid + it * 256;
      int tl = idx % G_TM, kk = idx / G_TM;
      float a = kv[(size_t)bb * CH * TT + (size_t)(kt * G_TK + kk) * TT + t0 + tl];
      As[tl * G_AP + kk] = (__bf16)a;
    }
#pragma unroll
    for (int it = 0; it < (G_TK * G_TN) / 256; ++it) {
      int idx = tid + it * 256;
      int nl = idx % G_TN, kk = idx / G_TN;
      float b = Wkv[(size_t)(kt * G_TK + kk) * (2 * CH) + n0 + nl];
      Bt[nl * G_BP + kk] = (__bf16)b;
    }
    if (kt + 1 < CH / G_TK)
      __builtin_prefetch(&kv[(size_t)bb * CH * TT +
                             (size_t)((kt + 1) * G_TK) * TT + t0 + tid], 0, 1);
    __syncthreads();
#pragma unroll
    for (int sm = 0; sm < 2; ++sm) {
      v16bf a = load_a_frag(As, G_AP, wm * 32 + sm * 16, 0, lane);
#pragma unroll
      for (int sn = 0; sn < 2; ++sn) {
        v16bf b = load_b_frag(Bt, G_BP, wn * 32 + sn * 16, 0, lane);
        acc[sm][sn] = wmma_bf16(a, b, acc[sm][sn]);
      }
    }
    __syncthreads();
  }

#pragma unroll
  for (int sm = 0; sm < 2; ++sm) {
#pragma unroll
    for (int sn = 0; sn < 2; ++sn) {
      int ng = n0 + wn * 32 + sn * 16 + (lane & 15);
      float bias = bkv[ng];
      __bf16* buf = (ng < CH) ? Kbuf : Vbuf;
      int nc = (ng < CH) ? ng : ng - CH;
#pragma unroll
      for (int r = 0; r < 8; ++r) {
        int m = m0 + wm * 32 + sm * 16 + r + 8 * (lane >> 4);
        buf[(size_t)m * CH + nc] = (__bf16)(acc[sm][sn][r] + bias);
      }
    }
  }
}

// ---------------------------------------------------------------------------
// Kernel 2: flash attention, software-pipelined.  K tiles stream via
// global_load_async_to_lds_b128 into a double buffer; V tiles prefetch into
// VGPRs and are transposed into LDS after compute.  Stage j+1 overlaps the
// WMMA/softmax work of stage j.
// ---------------------------------------------------------------------------
constexpr int QP = 72, KP = 72, VP = 72, PP = 72;
constexpr int NKB = TT / 64;

__global__ __launch_bounds__(128) void attn_kernel(
    const float* __restrict__ q, const __bf16* __restrict__ Kbuf,
    const __bf16* __restrict__ Vbuf, __bf16* __restrict__ Ybuf) {
  __shared__ __align__(16) __bf16 Qs[64 * QP];
  __shared__ __align__(16) __bf16 Ks[2][64 * KP];
  __shared__ __align__(16) __bf16 Vt[2][HD * VP];   // transposed [hd][key]
  __shared__ __align__(16) __bf16 Ps[4 * 16 * PP];  // per-wave P scratch

  const int tid = threadIdx.x, lane = tid & 31, w = tid >> 5;
  const int qb = blockIdx.x, hh = blockIdx.y, bb = blockIdx.z;
  const int qbase = qb * 64;

  const __bf16* Kb = Kbuf + (size_t)bb * TT * CH + hh * HD;
  const __bf16* Vb = Vbuf + (size_t)bb * TT * CH + hh * HD;

  // 64 keys x 64 ch = 512 16-byte chunks -> 4 per thread.
  auto issue_k = [&](int j, int b) {
#pragma unroll
    for (int i = 0; i < 4; ++i) {
      int id = i * 128 + tid;
      int key = id >> 3, c0 = (id & 7) * 8;
      async_load_b128(lds_off(&Ks[b][key * KP + c0]),
                      Kb + (size_t)(j * 64 + key) * CH + c0);
    }
  };
  auto load_v = [&](int j, v8bf* vr) {
#pragma unroll
    for (int i = 0; i < 4; ++i) {
      int id = i * 128 + tid;
      int key = id >> 3, c0 = (id & 7) * 8;
      vr[i] = *(const v8bf*)(Vb + (size_t)(j * 64 + key) * CH + c0);
    }
  };
  auto store_vt = [&](int b, const v8bf* vr) {
#pragma unroll
    for (int i = 0; i < 4; ++i) {
      int id = i * 128 + tid;
      int key = id >> 3, c0 = (id & 7) * 8;
#pragma unroll
      for (int e = 0; e < 8; ++e) Vt[b][(c0 + e) * VP + key] = vr[i][e];
    }
  };

  // Prologue: stage Q (f32 [c][t] -> bf16 [t][d]), K(0) async, V(0).
  for (int it = 0; it < (64 * HD) / 128; ++it) {
    int idx = tid + it * 128;
    int tl = idx % 64, d = idx / 64;
    float val = q[(size_t)bb * CH * TT + (size_t)(hh * HD + d) * TT + qbase + tl];
    Qs[tl * QP + d] = (__bf16)val;
  }
  v8bf vreg[4];
  issue_k(0, 0);
  load_v(0, vreg);
  store_vt(0, vreg);
  wait_asynccnt0();
  __syncthreads();

  v16bf Qa[2];
  Qa[0] = load_a_frag(Qs, QP, w * 16, 0, lane);
  Qa[1] = load_a_frag(Qs, QP, w * 16, 32, lane);

  float m_run[8], l_run[8];
  v8f Yacc[4];
  v8f zero = {};
#pragma unroll
  for (int r = 0; r < 8; ++r) { m_run[r] = -1e30f; l_run[r] = 0.0f; }
#pragma unroll
  for (int hb = 0; hb < 4; ++hb) Yacc[hb] = zero;

  for (int j = 0; j < NKB; ++j) {
    const int p = j & 1;
    const bool pre = (j + 1 < NKB);
    if (pre) {                       // kick off stage j+1 before compute
      issue_k(j + 1, p ^ 1);
      load_v(j + 1, vreg);
    }

    // S = Q K^T (contraction over hd=64, 2 WMMA K-steps), scaled 1/sqrt(64)
    v8f S[4];
#pragma unroll
    for (int nb = 0; nb < 4; ++nb) {
      v8f s = zero;
#pragma unroll
      for (int kk = 0; kk < 2; ++kk) {
        v16bf b = load_b_frag(&Ks[p][0], KP, nb * 16, kk * 32, lane);
        s = wmma_bf16(Qa[kk], b, s);
      }
#pragma unroll
      for (int r = 0; r < 8; ++r) s[r] *= 0.125f;
      S[nb] = s;
    }

    // Online softmax over 16-lane halves (row M = r + 8*(lane>>4))
    float mnew[8];
#pragma unroll
    for (int r = 0; r < 8; ++r) {
      float m = m_run[r];
#pragma unroll
      for (int nb = 0; nb < 4; ++nb) m = fmaxf(m, S[nb][r]);
#pragma unroll
      for (int mk = 1; mk <= 8; mk <<= 1)
        m = fmaxf(m, __shfl_xor(m, mk, 32));
      mnew[r] = m;
    }
    float lsum[8];
#pragma unroll
    for (int r = 0; r < 8; ++r) lsum[r] = 0.0f;
#pragma unroll
    for (int nb = 0; nb < 4; ++nb)
#pragma unroll
      for (int r = 0; r < 8; ++r) {
        float pexp = __expf(S[nb][r] - mnew[r]);
        S[nb][r] = pexp;
        lsum[r] += pexp;
      }
#pragma unroll
    for (int r = 0; r < 8; ++r) {
#pragma unroll
      for (int mk = 1; mk <= 8; mk <<= 1)
        lsum[r] += __shfl_xor(lsum[r], mk, 32);
      float sc = __expf(m_run[r] - mnew[r]);
      m_run[r] = mnew[r];
      l_run[r] = l_run[r] * sc + lsum[r];
#pragma unroll
      for (int hb = 0; hb < 4; ++hb) Yacc[hb][r] *= sc;
    }

    // P -> per-wave LDS (C-layout -> A-layout); LDS is in-order per wave.
    __bf16* Pw = Ps + w * 16 * PP;
#pragma unroll
    for (int nb = 0; nb < 4; ++nb)
#pragma unroll
      for (int r = 0; r < 8; ++r)
        Pw[(r + 8 * (lane >> 4)) * PP + nb * 16 + (lane & 15)] = (__bf16)S[nb][r];
    asm volatile("s_wait_dscnt 0" ::: "memory");

    // Y += P(16x64) V(64x64), contraction over keys
#pragma unroll
    for (int kk = 0; kk < 2; ++kk) {
      v16bf a = load_a_frag(Pw, PP, 0, kk * 32, lane);
#pragma unroll
      for (int hb = 0; hb < 4; ++hb) {
        v16bf b = load_b_frag(&Vt[p][0], VP, hb * 16, kk * 32, lane);
        Yacc[hb] = wmma_bf16(a, b, Yacc[hb]);
      }
    }

    if (pre) {                       // finish stage j+1 and publish it
      store_vt(p ^ 1, vreg);
      wait_asynccnt0();
      __syncthreads();
    }
  }

  // Normalize and store Y bf16 [bb][t][c]
#pragma unroll
  for (int hb = 0; hb < 4; ++hb)
#pragma unroll
    for (int r = 0; r < 8; ++r) {
      int trow = qbase + w * 16 + r + 8 * (lane >> 4);
      int cc = hh * HD + hb * 16 + (lane & 15);
      float y = Yacc[hb][r] / l_run[r];
      Ybuf[(size_t)bb * TT * CH + (size_t)trow * CH + cc] = (__bf16)y;
    }
}

// ---------------------------------------------------------------------------
// Kernel 3: output projection.  A-tile (bf16, 128 rows x 64B, stride 1KB) is
// DMA'd by the Tensor Data Mover straight into padded LDS (pad 4 DW every
// 16 DW reproduces the 80B pitch); one wave issues the D#, waits TENSORcnt.
// ---------------------------------------------------------------------------
__global__ __launch_bounds__(256) void proj_kernel(
    const __bf16* __restrict__ Ybuf, const float* __restrict__ Wp,
    const float* __restrict__ bp, float* __restrict__ out) {
  __shared__ __align__(16) __bf16 As[G_TM * G_AP];
  __shared__ __align__(16) __bf16 Bt[G_TN * G_BP];

  const int tid = threadIdx.x, lane = tid & 31, wave = tid >> 5;
  const int wm = wave & 3, wn = wave >> 2;
  const int m0 = blockIdx.x * G_TM;
  const int n0 = blockIdx.y * G_TN;

  v8f acc[2][2];
  v8f zero = {};
#pragma unroll
  for (int i = 0; i < 2; ++i)
#pragma unroll
    for (int j = 0; j < 2; ++j) acc[i][j] = zero;

  for (int kt = 0; kt < CH / G_TK; ++kt) {
    if (wave == 0) {
      // D# group0: count=1 | lds_addr | global_addr[56:0] | type=2
      unsigned long long ga =
          (unsigned long long)(const void*)(Ybuf + (size_t)m0 * CH + kt * G_TK);
      u32x4 g0 = {1u, lds_off(As), (unsigned)ga,
                  (unsigned)((ga >> 32) & 0x01FFFFFFull) | 0x80000000u};
      // D# group1: data_size=2B, pad_enable, pad_interval=16DW(64B),
      // pad_amount=4DW(16B); tensor_dim0=32, tensor_dim1=128, tile 32x128,
      // dim0_stride=512 elements.
      i32x8 g1 = {(1 << 16) | (1 << 20) | (3 << 22) | (3 << 25),
                  (G_TK << 16), (G_TM << 16), (G_TK << 16),
                  G_TM, CH, 0, 0};
      i32x4 gz = {0, 0, 0, 0};
#if defined(__clang_major__) && __clang_major__ >= 23
      i32x8 gz8 = {0, 0, 0, 0, 0, 0, 0, 0};
      __builtin_amdgcn_tensor_load_to_lds(g0, g1, gz, gz, gz8, 0);
#else
      __builtin_amdgcn_tensor_load_to_lds(g0, g1, gz, gz, 0);
#endif
    }
    // B stage (f32 -> bf16, transposed) on the VALU path, overlaps the DMA.
#pragma unroll
    for (int it = 0; it < (G_TK * G_TN) / 256; ++it) {
      int idx = tid + it * 256;
      int nl = idx % G_TN, kk = idx / G_TN;
      float b = Wp[(size_t)(kt * G_TK + kk) * CH + n0 + nl];
      Bt[nl * G_BP + kk] = (__bf16)b;
    }
    if (wave == 0) __builtin_amdgcn_s_wait_tensorcnt(0);
    __syncthreads();
#pragma unroll
    for (int sm = 0; sm < 2; ++sm) {
      v16bf a = load_a_frag(As, G_AP, wm * 32 + sm * 16, 0, lane);
#pragma unroll
      for (int sn = 0; sn < 2; ++sn) {
        v16bf b = load_b_frag(Bt, G_BP, wn * 32 + sn * 16, 0, lane);
        acc[sm][sn] = wmma_bf16(a, b, acc[sm][sn]);
      }
    }
    __syncthreads();
  }

#pragma unroll
  for (int sm = 0; sm < 2; ++sm) {
#pragma unroll
    for (int sn = 0; sn < 2; ++sn) {
      int ng = n0 + wn * 32 + sn * 16 + (lane & 15);
      float bias = bp[ng];
#pragma unroll
      for (int r = 0; r < 8; ++r) {
        int m = m0 + wm * 32 + sm * 16 + r + 8 * (lane >> 4);
        int bbv = m >> 10, t = m & (TT - 1);
        out[((size_t)bbv * CH + ng) * TT + t] = acc[sm][sn][r] + bias;
      }
    }
  }
}

extern "C" void kernel_launch(void* const* d_in, const int* in_sizes, int n_in,
                              void* d_out, int out_size, void* d_ws, size_t ws_size,
                              hipStream_t stream) {
  const float* q   = (const float*)d_in[0];
  const float* kv  = (const float*)d_in[1];
  const float* Wkv = (const float*)d_in[2];
  const float* bkv = (const float*)d_in[3];
  const float* Wp  = (const float*)d_in[4];
  const float* bp  = (const float*)d_in[5];
  float* out = (float*)d_out;

  const size_t elems = (size_t)NB * TT * CH;
  __bf16* Kbuf = (__bf16*)d_ws;
  __bf16* Vbuf = Kbuf + elems;
  __bf16* Ybuf = Vbuf + elems;

  kv_proj_kernel<<<dim3((NB * TT) / G_TM, (2 * CH) / G_TN), 256, 0, stream>>>(
      kv, Wkv, bkv, Kbuf, Vbuf);
  attn_kernel<<<dim3(TT / 64, NHEAD, NB), 128, 0, stream>>>(q, Kbuf, Vbuf, Ybuf);
  proj_kernel<<<dim3((NB * TT) / G_TM, CH / G_TN), 256, 0, stream>>>(
      Ybuf, Wp, bp, out);
}
